// CoarseMatching_7421703487994
// MI455X (gfx1250) — matrix operand order repaired and checked by reading, not compiled, not run on hardware
//
#include <hip/hip_runtime.h>

// ---------------------------------------------------------------------------
// CoarseMatching for MI455X (gfx1250, wave32, WMMA bf16 16x16x32)
//
// Pipeline (all on `stream`):
//   0) init_stats : zero row/col max(encoded) + sumexp accumulators
//   1) proj_kernel x2 : f = (feat @ W^T + b) / 16, stored bf16   (WMMA bf16)
//   2) sim_kernel<0>  : recompute sim tiles -> atomic row/col max
//   3) sim_kernel<1>  : recompute sim tiles -> atomic row/col sum(exp)
//   4) sim_kernel<2>  : recompute sim tiles -> conf01/conf10/mconf (NT stores)
// sim = 10 * (f0p . f1p)   (TEMPERATURE = 0.1)
// Deterministic recomputation => sim == rowmax reproduces argmax equality.
//
// CDNA5 specifics: v_wmma_f32_16x16x32_bf16 matrix cores; A/B tiles staged to
// LDS with global_load_async_to_lds_b128 (ASYNCcnt) + s_wait_asynccnt; padded
// LDS stride (528B) for conflict-free ds_load_b128 fragment reads; 553MB
// output streamed with non-temporal stores so L2 keeps the bf16 features.
// ---------------------------------------------------------------------------

#define L_DIM 4800
#define C_DIM 256
#define B_DIM 2
#define M_ROWS (B_DIM * L_DIM)     // 9600 flattened projection rows
#define SM_STRIDE 264              // 256 + 8 bf16 pad -> 528B row stride

typedef __attribute__((ext_vector_type(8)))  float  v8f;
typedef __attribute__((ext_vector_type(8)))  __bf16 v8bf;
typedef __attribute__((ext_vector_type(16))) __bf16 v16bf;

// ---- order-preserving float <-> u32 for atomicMax on floats ---------------
__device__ __forceinline__ unsigned fenc(float x) {
    unsigned u = __float_as_uint(x);
    return (u & 0x80000000u) ? ~u : (u | 0x80000000u);
}
__device__ __forceinline__ float fdec(unsigned u) {
    return __uint_as_float((u & 0x80000000u) ? (u & 0x7FFFFFFFu) : ~u);
}

// ---- WMMA wrapper ----------------------------------------------------------
__device__ __forceinline__ v8f wmma_bf16(v16bf a, v16bf b, v8f c) {
    return __builtin_amdgcn_wmma_f32_16x16x32_bf16(
        /*neg_a=*/false, a, /*neg_b=*/false, b,
        /*c_mod=*/(short)0, c, /*reuse_a=*/false, /*reuse_b=*/false);
}

// ---- fragment loads from LDS tiles (row stride SM_STRIDE bf16) ------------
// A 16x32 (MxK): lanes 0-15 M=0..15, V0..3 K=kb..kb+7, V4..7 K=16+kb..16+kb+7
__device__ __forceinline__ v16bf lds_a_frag(const __bf16* sm, int row, int k, int lane) {
    const int hl = lane >> 4, m = lane & 15;
    const __bf16* p = sm + (row + m) * SM_STRIDE + k + hl * 8;
    v8bf lo = *(const v8bf*)(p);
    v8bf hi = *(const v8bf*)(p + 16);
    return __builtin_shufflevector(lo, hi, 0,1,2,3,4,5,6,7,8,9,10,11,12,13,14,15);
}
// B 32x16 (KxN): B[k][n] = tile[n][k]; lanes 0-15 V0..3 K=kb..kb+7, V4..7 K=16+kb..16+kb+7 (kb=0/8)
__device__ __forceinline__ v16bf lds_b_frag(const __bf16* sm, int row, int k, int lane) {
    const int hl = lane >> 4, n = lane & 15;
    const __bf16* p = sm + (row + n) * SM_STRIDE + k + hl * 8;
    v8bf lo = *(const v8bf*)(p);
    v8bf hi = *(const v8bf*)(p + 16);
    return __builtin_shufflevector(lo, hi, 0,1,2,3,4,5,6,7,8,9,10,11,12,13,14,15);
}

// f32 sources with on-the-fly bf16 conversion (projection inputs, global)
__device__ __forceinline__ v16bf load_a_f32(const float* base, int row, int k, int lane) {
    const int hl = lane >> 4, m = lane & 15;
    const float* p = base + (size_t)(row + m) * C_DIM + k + hl * 8;
    float4 x0 = *(const float4*)(p);
    float4 x1 = *(const float4*)(p + 4);
    float4 y0 = *(const float4*)(p + 16);
    float4 y1 = *(const float4*)(p + 20);
    v16bf a;
    a[0]=(__bf16)x0.x; a[1]=(__bf16)x0.y; a[2]=(__bf16)x0.z; a[3]=(__bf16)x0.w;
    a[4]=(__bf16)x1.x; a[5]=(__bf16)x1.y; a[6]=(__bf16)x1.z; a[7]=(__bf16)x1.w;
    a[8]=(__bf16)y0.x; a[9]=(__bf16)y0.y; a[10]=(__bf16)y0.z; a[11]=(__bf16)y0.w;
    a[12]=(__bf16)y1.x; a[13]=(__bf16)y1.y; a[14]=(__bf16)y1.z; a[15]=(__bf16)y1.w;
    return a;
}
__device__ __forceinline__ v16bf load_b_f32(const float* base, int row, int k, int lane) {
    const int hl = lane >> 4, n = lane & 15;
    const float* p = base + (size_t)(row + n) * C_DIM + k + hl * 8;
    float4 x0 = *(const float4*)(p);
    float4 x1 = *(const float4*)(p + 4);
    float4 x2 = *(const float4*)(p + 16);
    float4 x3 = *(const float4*)(p + 20);
    v16bf b;
    b[0]=(__bf16)x0.x; b[1]=(__bf16)x0.y; b[2]=(__bf16)x0.z; b[3]=(__bf16)x0.w;
    b[4]=(__bf16)x1.x; b[5]=(__bf16)x1.y; b[6]=(__bf16)x1.z; b[7]=(__bf16)x1.w;
    b[8]=(__bf16)x2.x; b[9]=(__bf16)x2.y; b[10]=(__bf16)x2.z; b[11]=(__bf16)x2.w;
    b[12]=(__bf16)x3.x; b[13]=(__bf16)x3.y; b[14]=(__bf16)x3.z; b[15]=(__bf16)x3.w;
    return b;
}

// ---- async global -> LDS copy of one 16B chunk (CDNA5, ASYNCcnt) ----------
__device__ __forceinline__ void async_copy_b128(unsigned lds_addr, const void* gaddr) {
    asm volatile("global_load_async_to_lds_b128 %0, %1, off"
                 :: "v"(lds_addr), "v"(gaddr) : "memory");
}
__device__ __forceinline__ void wait_async_zero() {
    asm volatile("s_wait_asynccnt 0x0" ::: "memory");
}

// ---------------------------------------------------------------------------
// init: zero the stat accumulators (rowmax/colmax encoded, rowsum/colsum)
__global__ void init_stats(unsigned* __restrict__ rmax, unsigned* __restrict__ cmax,
                           float* __restrict__ rsum, float* __restrict__ csum, int n) {
    int i = blockIdx.x * blockDim.x + threadIdx.x;
    if (i < n) { rmax[i] = 0u; cmax[i] = 0u; rsum[i] = 0.0f; csum[i] = 0.0f; }
}

// ---------------------------------------------------------------------------
// projection: dst[m, c] = bf16( (feat[m,:] . W[c,:] + bias[c]) / 16 )
// grid: (256/64, 9600/64) = (4, 150), block = 128 (4 waves, each 32x32)
__global__ __launch_bounds__(128) void proj_kernel(
    const float* __restrict__ feat, const float* __restrict__ W,
    const float* __restrict__ bias, __bf16* __restrict__ dst) {
    const int lane = threadIdx.x & 31;
    const int wave = threadIdx.x >> 5;
    const int m0 = blockIdx.y * 64 + (wave >> 1) * 32;
    const int n0 = blockIdx.x * 64 + (wave & 1) * 32;

    v8f acc[2][2] = {};
#pragma unroll
    for (int ks = 0; ks < C_DIM; ks += 32) {
        v16bf a0 = load_a_f32(feat, m0,      ks, lane);
        v16bf a1 = load_a_f32(feat, m0 + 16, ks, lane);
        v16bf b0 = load_b_f32(W,    n0,      ks, lane);
        v16bf b1 = load_b_f32(W,    n0 + 16, ks, lane);
        acc[0][0] = wmma_bf16(a0, b0, acc[0][0]);
        acc[0][1] = wmma_bf16(a0, b1, acc[0][1]);
        acc[1][0] = wmma_bf16(a1, b0, acc[1][0]);
        acc[1][1] = wmma_bf16(a1, b1, acc[1][1]);
    }
    const int hl = lane >> 4, lm = lane & 15;
#pragma unroll
    for (int mi = 0; mi < 2; ++mi)
#pragma unroll
        for (int nj = 0; nj < 2; ++nj) {
            const int col = n0 + nj * 16 + lm;
            const float bc = bias[col];
#pragma unroll
            for (int r = 0; r < 8; ++r) {
                const int row = m0 + mi * 16 + hl * 8 + r;
                float v = (acc[mi][nj][r] + bc) * 0.0625f;   // / sqrt(256)
                dst[(size_t)row * C_DIM + col] = (__bf16)v;
            }
        }
}

// ---------------------------------------------------------------------------
__device__ __forceinline__ bool interior(int i, int h, int w) {
    int r = i / w, c = i - r * w;
    return (r >= 2) & (r < h - 2) & (c >= 2) & (c < w - 2);
}

// sim passes. MODE: 0=row/col max, 1=row/col sum(exp), 2=final outputs.
// grid: (75, 75, 2), block = 128 (4 waves, each 32x32 of the 64x64 tile)
template <int MODE>
__global__ __launch_bounds__(128) void sim_kernel(
    const __bf16* __restrict__ f0b, const __bf16* __restrict__ f1b,
    unsigned* __restrict__ rmaxu, unsigned* __restrict__ cmaxu,
    float* __restrict__ rsum, float* __restrict__ csum,
    float* __restrict__ out,
    const int* __restrict__ h0p, const int* __restrict__ w0p,
    const int* __restrict__ h1p, const int* __restrict__ w1p) {
    __shared__ __bf16 smA[64 * SM_STRIDE];
    __shared__ __bf16 smB[64 * SM_STRIDE];

    const int lane = threadIdx.x & 31;
    const int wave = threadIdx.x >> 5;
    const int bb = blockIdx.z;
    const int m0wg = blockIdx.y * 64;
    const int n0wg = blockIdx.x * 64;
    const __bf16* A  = f0b + (size_t)bb * L_DIM * C_DIM + (size_t)m0wg * C_DIM;
    const __bf16* Bm = f1b + (size_t)bb * L_DIM * C_DIM + (size_t)n0wg * C_DIM;

    // ---- async-stage both 64x256 bf16 tiles into LDS (padded rows) --------
    {
        const unsigned baseA = (unsigned)(uintptr_t)(&smA[0]);
        const unsigned baseB = (unsigned)(uintptr_t)(&smB[0]);
        // 64 rows x 512B = 2048 x 16B chunks per tile; 128 threads x 16 chunks
        for (int c = threadIdx.x; c < 2048; c += 128) {
            const int row = c >> 5;            // 32 chunks per 512B row
            const int col = c & 31;
            const unsigned lofs = (unsigned)(row * (SM_STRIDE * 2) + col * 16);
            const char* gA = (const char*)A  + (size_t)row * 512 + col * 16;
            const char* gB = (const char*)Bm + (size_t)row * 512 + col * 16;
            async_copy_b128(baseA + lofs, gA);
            async_copy_b128(baseB + lofs, gB);
        }
        wait_async_zero();
        __syncthreads();
    }

    const int mw = (wave >> 1) * 32;           // wave tile inside 64x64
    const int nw = (wave & 1) * 32;
    const int m0 = m0wg + mw;
    const int n0 = n0wg + nw;

    v8f acc[2][2] = {};
#pragma unroll
    for (int ks = 0; ks < C_DIM; ks += 32) {
        v16bf a0 = lds_a_frag(smA, mw,      ks, lane);
        v16bf a1 = lds_a_frag(smA, mw + 16, ks, lane);
        v16bf b0 = lds_b_frag(smB, nw,      ks, lane);
        v16bf b1 = lds_b_frag(smB, nw + 16, ks, lane);
        acc[0][0] = wmma_bf16(a0, b0, acc[0][0]);
        acc[0][1] = wmma_bf16(a0, b1, acc[0][1]);
        acc[1][0] = wmma_bf16(a1, b0, acc[1][0]);
        acc[1][1] = wmma_bf16(a1, b1, acc[1][1]);
    }
    const int hl = lane >> 4, lm = lane & 15;
    const int rbase = bb * L_DIM;

    if (MODE == 0) {
        // ---- row max (reduce across N=lane within each 16-lane half) ------
#pragma unroll
        for (int mi = 0; mi < 2; ++mi)
#pragma unroll
            for (int r = 0; r < 8; ++r) {
                float v = fmaxf(acc[mi][0][r], acc[mi][1][r]) * 10.0f;
                v = fmaxf(v, __shfl_xor(v, 1, 32));
                v = fmaxf(v, __shfl_xor(v, 2, 32));
                v = fmaxf(v, __shfl_xor(v, 4, 32));
                v = fmaxf(v, __shfl_xor(v, 8, 32));
                if (lm == 0)
                    atomicMax(&rmaxu[rbase + m0 + mi * 16 + hl * 8 + r], fenc(v));
            }
        // ---- col max (reduce across M: regs, M-tiles, then halves) --------
#pragma unroll
        for (int nj = 0; nj < 2; ++nj) {
            float v = -3.0e38f;
#pragma unroll
            for (int mi = 0; mi < 2; ++mi)
#pragma unroll
                for (int r = 0; r < 8; ++r) v = fmaxf(v, acc[mi][nj][r]);
            v *= 10.0f;
            v = fmaxf(v, __shfl_xor(v, 16, 32));
            if (lane < 16)
                atomicMax(&cmaxu[rbase + n0 + nj * 16 + lane], fenc(v));
        }
    } else if (MODE == 1) {
        // ---- row sum of exp(sim - rowmax) ---------------------------------
#pragma unroll
        for (int mi = 0; mi < 2; ++mi)
#pragma unroll
            for (int r = 0; r < 8; ++r) {
                const int row = m0 + mi * 16 + hl * 8 + r;
                const float rm = fdec(rmaxu[rbase + row]);
                float e = __expf(acc[mi][0][r] * 10.0f - rm) +
                          __expf(acc[mi][1][r] * 10.0f - rm);
                e += __shfl_xor(e, 1, 32);
                e += __shfl_xor(e, 2, 32);
                e += __shfl_xor(e, 4, 32);
                e += __shfl_xor(e, 8, 32);
                if (lm == 0) atomicAdd(&rsum[rbase + row], e);
            }
        // ---- col sum of exp(sim - colmax) ---------------------------------
#pragma unroll
        for (int nj = 0; nj < 2; ++nj) {
            const int col = n0 + nj * 16 + lm;
            const float cm = fdec(cmaxu[rbase + col]);
            float e = 0.0f;
#pragma unroll
            for (int mi = 0; mi < 2; ++mi)
#pragma unroll
                for (int r = 0; r < 8; ++r)
                    e += __expf(acc[mi][nj][r] * 10.0f - cm);
            e += __shfl_xor(e, 16, 32);
            if (lane < 16) atomicAdd(&csum[rbase + n0 + nj * 16 + lane], e);
        }
    } else {
        // ---- final: conf01 / conf10 / mconf, streamed past L2 -------------
        const int h0 = *h0p, w0 = *w0p, h1 = *h1p, w1 = *w1p;
        const size_t PS = (size_t)B_DIM * L_DIM * L_DIM;     // one output plane
        float cm[2], csv[2];
        bool int1[2];
#pragma unroll
        for (int nj = 0; nj < 2; ++nj) {
            const int col = n0 + nj * 16 + lm;
            cm[nj]  = fdec(cmaxu[rbase + col]);
            csv[nj] = 1.0f / csum[rbase + col];
            int1[nj] = interior(col, h1, w1);
        }
#pragma unroll
        for (int mi = 0; mi < 2; ++mi)
#pragma unroll
            for (int r = 0; r < 8; ++r) {
                const int row = m0 + mi * 16 + hl * 8 + r;
                const float rm  = fdec(rmaxu[rbase + row]);
                const float rsv = 1.0f / rsum[rbase + row];
                const bool int0 = interior(row, h0, w0);
#pragma unroll
                for (int nj = 0; nj < 2; ++nj) {
                    const int col = n0 + nj * 16 + lm;
                    const float s = acc[mi][nj][r] * 10.0f;
                    const float c01 = __expf(s - rm) * rsv;
                    const float c10 = __expf(s - cm[nj]) * csv[nj];
                    const bool m01 = (c01 > 0.2f) && (s == rm);
                    const bool m10 = (c10 > 0.2f) && (s == cm[nj]);
                    const bool msk = (m01 | m10) && int0 && int1[nj];
                    const float mc = msk ? fmaxf(c01, c10) : 0.0f;
                    const size_t idx = (size_t)bb * L_DIM * L_DIM +
                                       (size_t)row * L_DIM + col;
                    __builtin_nontemporal_store(c01, &out[idx]);
                    __builtin_nontemporal_store(c10, &out[PS + idx]);
                    __builtin_nontemporal_store(mc,  &out[2 * PS + idx]);
                }
            }
    }
}

// ---------------------------------------------------------------------------
extern "C" void kernel_launch(void* const* d_in, const int* in_sizes, int n_in,
                              void* d_out, int out_size, void* d_ws, size_t ws_size,
                              hipStream_t stream) {
    const float* feat0 = (const float*)d_in[0];
    const float* feat1 = (const float*)d_in[1];
    const float* W     = (const float*)d_in[2];
    const float* bias  = (const float*)d_in[3];
    const int*   h0p   = (const int*)d_in[4];
    const int*   w0p   = (const int*)d_in[5];
    const int*   h1p   = (const int*)d_in[6];
    const int*   w1p   = (const int*)d_in[7];
    float* out = (float*)d_out;

    char* ws = (char*)d_ws;
    const size_t FB = (size_t)M_ROWS * C_DIM * sizeof(__bf16);  // 4,915,200 B
    __bf16*   f0b  = (__bf16*)(ws);
    __bf16*   f1b  = (__bf16*)(ws + FB);
    unsigned* rmax = (unsigned*)(ws + 2 * FB);
    unsigned* cmax = (unsigned*)(ws + 2 * FB + (size_t)M_ROWS * 4);
    float*    rsum = (float*)   (ws + 2 * FB + (size_t)M_ROWS * 8);
    float*    csum = (float*)   (ws + 2 * FB + (size_t)M_ROWS * 12);

    init_stats<<<(M_ROWS + 255) / 256, 256, 0, stream>>>(rmax, cmax, rsum, csum, M_ROWS);

    dim3 pgrid(C_DIM / 64, M_ROWS / 64);           // (4, 150)
    proj_kernel<<<pgrid, 128, 0, stream>>>(feat0, W, bias, f0b);
    proj_kernel<<<pgrid, 128, 0, stream>>>(feat1, W, bias, f1b);

    dim3 sgrid(L_DIM / 64, L_DIM / 64, B_DIM);     // (75, 75, 2)
    sim_kernel<0><<<sgrid, 128, 0, stream>>>(f0b, f1b, rmax, cmax, rsum, csum, out,
                                             h0p, w0p, h1p, w1p);
    sim_kernel<1><<<sgrid, 128, 0, stream>>>(f0b, f1b, rmax, cmax, rsum, csum, out,
                                             h0p, w0p, h1p, w1p);
    sim_kernel<2><<<sgrid, 128, 0, stream>>>(f0b, f1b, rmax, cmax, rsum, csum, out,
                                             h0p, w0p, h1p, w1p);
}
